// GETModel_12635793785238
// MI455X (gfx1250) — compile-verified, matrix operand order checked by reading, not run on hardware
//
#include <hip/hip_runtime.h>
#include <hip/hip_bf16.h>
#include <stdint.h>

// ---------------- problem constants ----------------
#define N_NODES 40000
#define N_EDGES 160000
#define DDIM    128
#define NHEAD   4
#define HDIM    32
#define RRANK   4
#define KSLOT   32
#define FTOT    1408          // 128+128+512+512+128 concatenated projections
#define OFF_Q2  0
#define OFF_K2  128
#define OFF_Q3  256
#define OFF_K3  768
#define OFF_QM  1280

#define INV_SQRT_HD   0.17677669529663687f   // 1/sqrt(32)
#define INV_SQRT_RHD  0.08838834764831845f   // 1/sqrt(128)
#define ENC_NEG_INF   0x007FFFFFu            // monotonic encoding of -inf

typedef __bf16 bf16;
typedef __attribute__((ext_vector_type(16))) __bf16 v16bf;
typedef __attribute__((ext_vector_type(8)))  __bf16 v8bf;
typedef __attribute__((ext_vector_type(8)))  float  v8f;

// ---------------- device helpers ----------------
__device__ __forceinline__ float softplusf(float x) {
    return (x > 20.f) ? x : log1pf(expf(x));
}
__device__ __forceinline__ unsigned bf16bits(float f) {
    unsigned u = __float_as_uint(f);
    return (u + 0x7FFFu + ((u >> 16) & 1u)) >> 16;   // round-to-nearest-even
}
// monotonic float<->uint mapping for atomicMax on floats
__device__ __forceinline__ unsigned fenc(float f) {
    unsigned u = __float_as_uint(f);
    return (u & 0x80000000u) ? ~u : (u | 0x80000000u);
}
__device__ __forceinline__ float fdec(unsigned e) {
    return (e & 0x80000000u) ? __uint_as_float(e & 0x7FFFFFFFu)
                             : __uint_as_float(~e);
}
__device__ __forceinline__ float wave_sum(float v) {
#pragma unroll
    for (int o = 16; o > 0; o >>= 1) v += __shfl_xor(v, o, 32);
    return v;
}

// ---------------- weight prep ----------------
// Wcat  [FTOT x 128] bf16 (row-major) and WcatT [128 x FTOT] bf16
__global__ void k_build_wcat(const float* __restrict__ WQ2, const float* __restrict__ WK2,
                             const float* __restrict__ WQ3, const float* __restrict__ WK3,
                             const float* __restrict__ WQm,
                             unsigned short* __restrict__ wcat,
                             unsigned short* __restrict__ wcatT, int total) {
    int idx = blockIdx.x * blockDim.x + threadIdx.x;
    if (idx >= total) return;
    int f = idx / DDIM, d = idx - f * DDIM;
    float v;
    if      (f <  128) v = WQ2[(f        ) * DDIM + d];
    else if (f <  256) v = WK2[(f -  128 ) * DDIM + d];
    else if (f <  768) v = WQ3[(f -  256 ) * DDIM + d];
    else if (f < 1280) v = WK3[(f -  768 ) * DDIM + d];
    else               v = WQm[(f - 1280 ) * DDIM + d];
    unsigned short b = (unsigned short)bf16bits(v);
    wcat [(size_t)f * DDIM + d] = b;
    wcatT[(size_t)d * FTOT + f] = b;
}

// Km[h][k][z] = sum_d B_mem[k,d] * W_Km[h,z,d]
__global__ void k_build_km(const float* __restrict__ Bmem, const float* __restrict__ WKm,
                           float* __restrict__ Km, int total) {
    int idx = blockIdx.x * blockDim.x + threadIdx.x;
    if (idx >= total) return;
    int z = idx & (HDIM - 1);
    int k = (idx >> 5) & (KSLOT - 1);
    int h = idx >> 10;
    float a = 0.f;
    for (int d = 0; d < DDIM; ++d)
        a += Bmem[k * DDIM + d] * WKm[((h * HDIM + z) * DDIM) + d];
    Km[idx] = a;
}

// ---------------- LayerNorm forward ----------------
__global__ void k_ln_fwd(const float* __restrict__ G, const float* __restrict__ gamma,
                         const float* __restrict__ beta,
                         unsigned short* __restrict__ gbf,
                         float* __restrict__ mu_out, float* __restrict__ rs_out, int n) {
    int node = (blockIdx.x * blockDim.x + threadIdx.x) >> 5;
    int lane = threadIdx.x & 31;
    if (node >= n) return;
    int d = lane * 4;
    float4 x = *(const float4*)(G + (size_t)node * DDIM + d);
    float s  = wave_sum(x.x + x.y + x.z + x.w);
    float ss = wave_sum(x.x * x.x + x.y * x.y + x.z * x.z + x.w * x.w);
    float mu  = s * (1.f / DDIM);
    float var = ss * (1.f / DDIM) - mu * mu;
    float rs  = rsqrtf(var + 1e-5f);
    if (lane == 0) { mu_out[node] = mu; rs_out[node] = rs; }
    float g0 = (x.x - mu) * rs * gamma[d + 0] + beta[d + 0];
    float g1 = (x.y - mu) * rs * gamma[d + 1] + beta[d + 1];
    float g2 = (x.z - mu) * rs * gamma[d + 2] + beta[d + 2];
    float g3 = (x.w - mu) * rs * gamma[d + 3] + beta[d + 3];
    uint2 pk;
    pk.x = bf16bits(g0) | (bf16bits(g1) << 16);
    pk.y = bf16bits(g2) | (bf16bits(g3) << 16);
    *(uint2*)(gbf + (size_t)node * DDIM + d) = pk;
}

// ---------------- WMMA GEMM: C[M x Nc] = A[M x Kd] * B[Nc x Kd]^T ----------------
// one wave per 16(M) x 64(N) output macro-tile: A loaded once per K-step,
// reused across 4 B tiles -> 4 wmma per K-step, 4x less A traffic.
__global__ void k_gemm_bf16(const bf16* __restrict__ A, int lda,
                            const bf16* __restrict__ B, int ldb,
                            float* __restrict__ C, int ldc, int Kd) {
    const int lane = threadIdx.x & 31;
    const int half = lane >> 4;     // K-half select
    const int r16  = lane & 15;     // row (A) / col (B)
    const int m0 = blockIdx.x * 16;
    const int n0 = blockIdx.y * 64;
    const bf16* arow = A + (size_t)(m0 + r16) * lda;
    const bf16* brow = B + (size_t)(n0 + r16) * ldb;
    v8f acc[4] = {};
    for (int kk = 0; kk < Kd; kk += 32) {
        // 16-bit A 16x32 layout: VGPR0-3 hold K = half*8 + 0..7, VGPR4-7 hold K = 16 + half*8 + 0..7
        v8bf a0 = *(const v8bf*)(arow + kk + half * 8);
        v8bf a1 = *(const v8bf*)(arow + kk + 16 + half * 8);
        v16bf a;
#pragma unroll
        for (int i = 0; i < 8; ++i) { a[i] = a0[i]; a[i + 8] = a1[i]; }
#pragma unroll
        for (int t = 0; t < 4; ++t) {
            const bf16* br = brow + (size_t)(t * 16) * ldb;
            v8bf b0 = *(const v8bf*)(br + kk + half * 8);
            v8bf b1 = *(const v8bf*)(br + kk + 16 + half * 8);
            v16bf b;
#pragma unroll
            for (int i = 0; i < 8; ++i) { b[i] = b0[i]; b[i + 8] = b1[i]; }
            acc[t] = __builtin_amdgcn_wmma_f32_16x16x32_bf16(false, a, false, b,
                                                             (short)0, acc[t], false, false);
        }
    }
    // C/D layout: VGPR v, lanes 0-15 -> M=v, lanes 16-31 -> M=v+8; N = lane%16
#pragma unroll
    for (int t = 0; t < 4; ++t) {
        float* crow = C + (size_t)(m0 + 8 * half) * ldc + n0 + t * 16 + r16;
#pragma unroll
        for (int v = 0; v < 8; ++v) crow[(size_t)v * ldc] = acc[t][v];
    }
}

// ---------------- edge logits ----------------
__global__ void k_edge_logits(const float* __restrict__ P, const int* __restrict__ c2,
                              const int* __restrict__ u2, const int* __restrict__ mt,
                              const float* __restrict__ Ttau,
                              float* __restrict__ l2, float* __restrict__ l3, int total) {
    int idx = blockIdx.x * blockDim.x + threadIdx.x;
    if (idx >= total) return;
    int h = idx & 3, e = idx >> 2;
    int c = c2[e], u = u2[e], t = mt[e];
    const float* q2 = P + (size_t)c * FTOT + OFF_Q2 + h * HDIM;
    const float* k2 = P + (size_t)u * FTOT + OFF_K2 + h * HDIM;
    float a = 0.f;
#pragma unroll
    for (int z = 0; z < HDIM; ++z) a += q2[z] * k2[z];
    l2[idx] = a * INV_SQRT_HD;
    const float* q3 = P + (size_t)c * FTOT + OFF_Q3 + h * RRANK * HDIM;
    const float* k3 = P + (size_t)u * FTOT + OFF_K3 + h * RRANK * HDIM;
    const float* te = Ttau + ((size_t)t * NHEAD + h) * RRANK * HDIM;
    float b = 0.f;
    for (int j = 0; j < RRANK * HDIM; ++j) b += q3[j] * te[j] * k3[j];
    l3[idx] = b * INV_SQRT_RHD;
}

// ---------------- segment init / max / sum ----------------
__global__ void k_init(unsigned* __restrict__ m2, float* __restrict__ s2,
                       unsigned* __restrict__ m3, float* __restrict__ s3,
                       float* __restrict__ gn, int total) {
    int idx = blockIdx.x * blockDim.x + threadIdx.x;
    if (idx >= total) return;
    m2[idx] = ENC_NEG_INF; m3[idx] = ENC_NEG_INF;
    s2[idx] = 0.f;         s3[idx] = 0.f;
    if (idx == 0) gn[0] = 0.f;
}

__global__ void k_segmax(const float* __restrict__ l2, const float* __restrict__ l3,
                         const int* __restrict__ u2,
                         unsigned* __restrict__ m2, unsigned* __restrict__ m3,
                         const float* b2r, const float* b3r, int total) {
    int idx = blockIdx.x * blockDim.x + threadIdx.x;
    if (idx >= total) return;
    int h = idx & 3, e = idx >> 2;
    int seg = u2[e] * NHEAD + h;
    float beta2 = fminf(softplusf(*b2r), 5.f);
    float beta3 = fminf(softplusf(*b3r), 5.f);
    atomicMax(&m2[seg], fenc(beta2 * l2[idx]));
    atomicMax(&m3[seg], fenc(beta3 * l3[idx]));
}

__global__ void k_segsum(const float* __restrict__ l2, const float* __restrict__ l3,
                         const int* __restrict__ u2,
                         const unsigned* __restrict__ m2, const unsigned* __restrict__ m3,
                         float* __restrict__ s2, float* __restrict__ s3,
                         const float* b2r, const float* b3r, int total) {
    int idx = blockIdx.x * blockDim.x + threadIdx.x;
    if (idx >= total) return;
    int h = idx & 3, e = idx >> 2;
    int seg = u2[e] * NHEAD + h;
    float beta2 = fminf(softplusf(*b2r), 5.f);
    float beta3 = fminf(softplusf(*b3r), 5.f);
    atomicAdd(&s2[seg], expf(beta2 * l2[idx] - fdec(m2[seg])));
    atomicAdd(&s3[seg], expf(beta3 * l3[idx] - fdec(m3[seg])));
}

// ---------------- memory-slot backward: dQm -> dP[:,1280:1408] ----------------
__global__ void k_mem_bwd(const float* __restrict__ P, const float* __restrict__ Km,
                          float* __restrict__ dP,
                          const float* lammr, const float* betamr, int total) {
    int idx = blockIdx.x * blockDim.x + threadIdx.x;
    if (idx >= total) return;
    int h = idx & 3, n = idx >> 2;
    float lamm  = softplusf(*lammr);
    float betam = fminf(softplusf(*betamr), 5.f);
    const float* q  = P  + (size_t)n * FTOT + OFF_QM + h * HDIM;
    const float* km = Km + (size_t)h * KSLOT * HDIM;
    float p[KSLOT];
    float mx = -3.4e38f;
    for (int k = 0; k < KSLOT; ++k) {
        float d = 0.f;
#pragma unroll
        for (int z = 0; z < HDIM; ++z) d += q[z] * km[k * HDIM + z];
        p[k] = betam * d * INV_SQRT_HD;
        mx = fmaxf(mx, p[k]);
    }
    float s = 0.f;
    for (int k = 0; k < KSLOT; ++k) { p[k] = expf(p[k] - mx); s += p[k]; }
    float inv = 1.f / s;
    float* dq = dP + (size_t)n * FTOT + OFF_QM + h * HDIM;
    for (int z = 0; z < HDIM; ++z) {
        float a = 0.f;
        for (int k = 0; k < KSLOT; ++k) a += p[k] * km[k * HDIM + z];
        dq[z] = -lamm * INV_SQRT_HD * inv * a;
    }
}

// ---------------- edge backward: scatter into dP ----------------
__global__ void k_edge_bwd(const float* __restrict__ P,
                           const float* __restrict__ l2, const float* __restrict__ l3,
                           const unsigned* __restrict__ m2, const float* __restrict__ s2,
                           const unsigned* __restrict__ m3, const float* __restrict__ s3,
                           const int* __restrict__ c2, const int* __restrict__ u2,
                           const int* __restrict__ mt, const float* __restrict__ Ttau,
                           float* __restrict__ dP,
                           const float* lam2r, const float* lam3r,
                           const float* b2r, const float* b3r, int total) {
    int idx = blockIdx.x * blockDim.x + threadIdx.x;
    if (idx >= total) return;
    int s = idx & 31;
    int h = (idx >> 5) & 3;
    int e = idx >> 7;
    int c = c2[e], u = u2[e], t = mt[e];
    int seg = u * NHEAD + h;
    int eh  = e * NHEAD + h;
    float lam2  = softplusf(*lam2r);
    float lam3  = softplusf(*lam3r);
    float beta2 = fminf(softplusf(*b2r), 5.f);
    float beta3 = fminf(softplusf(*b3r), 5.f);
    // pairwise term
    float p2 = expf(beta2 * l2[eh] - fdec(m2[seg])) / s2[seg];
    float w2 = -lam2 * p2 * INV_SQRT_HD;
    atomicAdd(&dP[(size_t)c * FTOT + OFF_Q2 + h * HDIM + s],
              w2 * P[(size_t)u * FTOT + OFF_K2 + h * HDIM + s]);
    atomicAdd(&dP[(size_t)u * FTOT + OFF_K2 + h * HDIM + s],
              w2 * P[(size_t)c * FTOT + OFF_Q2 + h * HDIM + s]);
    // motif third-order term
    float p3 = expf(beta3 * l3[eh] - fdec(m3[seg])) / s3[seg];
    float w3 = -lam3 * p3 * INV_SQRT_RHD;
    const float* te = Ttau + ((size_t)t * NHEAD + h) * RRANK * HDIM;
#pragma unroll
    for (int r = 0; r < RRANK; ++r) {
        int j = r * HDIM + s;
        float tv = te[j];
        atomicAdd(&dP[(size_t)c * FTOT + OFF_Q3 + h * RRANK * HDIM + j],
                  w3 * tv * P[(size_t)u * FTOT + OFF_K3 + h * RRANK * HDIM + j]);
        atomicAdd(&dP[(size_t)u * FTOT + OFF_K3 + h * RRANK * HDIM + j],
                  w3 * tv * P[(size_t)c * FTOT + OFF_Q3 + h * RRANK * HDIM + j]);
    }
}

// ---------------- fp32 -> bf16 bulk convert ----------------
__global__ void k_cvt_bf16(const float* __restrict__ src, unsigned short* __restrict__ dst,
                           int total) {
    int idx = blockIdx.x * blockDim.x + threadIdx.x;
    if (idx >= total) return;
    dst[idx] = (unsigned short)bf16bits(src[idx]);
}

// ---------------- LayerNorm backward + global grad-norm ----------------
__global__ void k_ln_bwd(const float* __restrict__ G, const float* __restrict__ gamma,
                         const float* __restrict__ mu, const float* __restrict__ rs,
                         const float* __restrict__ dg, float* __restrict__ grad,
                         float* __restrict__ gn, int n) {
    int node = (blockIdx.x * blockDim.x + threadIdx.x) >> 5;
    int lane = threadIdx.x & 31;
    if (node >= n) return;
    int d = lane * 4;
    float4 x  = *(const float4*)(G  + (size_t)node * DDIM + d);
    float4 dv = *(const float4*)(dg + (size_t)node * DDIM + d);
    float m = mu[node], r = rs[node];
    float dy0 = dv.x * gamma[d + 0], dy1 = dv.y * gamma[d + 1];
    float dy2 = dv.z * gamma[d + 2], dy3 = dv.w * gamma[d + 3];
    float gh0 = (x.x - m) * r, gh1 = (x.y - m) * r;
    float gh2 = (x.z - m) * r, gh3 = (x.w - m) * r;
    float s1 = wave_sum(dy0 + dy1 + dy2 + dy3) * (1.f / DDIM);
    float s2 = wave_sum(dy0 * gh0 + dy1 * gh1 + dy2 * gh2 + dy3 * gh3) * (1.f / DDIM);
    float g0 = r * (dy0 - s1 - gh0 * s2);
    float g1 = r * (dy1 - s1 - gh1 * s2);
    float g2 = r * (dy2 - s1 - gh2 * s2);
    float g3 = r * (dy3 - s1 - gh3 * s2);
    *(float4*)(grad + (size_t)node * DDIM + d) = make_float4(g0, g1, g2, g3);
    float sq = wave_sum(g0 * g0 + g1 * g1 + g2 * g2 + g3 * g3);
    if (lane == 0) atomicAdd(gn, sq);
}

// ---------------- clipped update ----------------
__global__ void k_update(const float* __restrict__ G, const float* __restrict__ grad,
                         const float* __restrict__ gn, float* __restrict__ out, int n) {
    int node = (blockIdx.x * blockDim.x + threadIdx.x) >> 5;
    int lane = threadIdx.x & 31;
    if (node >= n) return;
    int d = lane * 4;
    float gnorm = sqrtf(gn[0] + 1e-12f);
    float gs = fminf(1.f, 1.f / (gnorm + 1e-6f));     // GRAD_CLIP = 1
    float4 x = *(const float4*)(G    + (size_t)node * DDIM + d);
    float4 g = *(const float4*)(grad + (size_t)node * DDIM + d);
    float n0 = x.x - 0.9999f * g.x * gs;
    float n1 = x.y - 0.9999f * g.y * gs;
    float n2 = x.z - 0.9999f * g.z * gs;
    float n3 = x.w - 0.9999f * g.w * gs;
    float rn = sqrtf(wave_sum(n0 * n0 + n1 * n1 + n2 * n2 + n3 * n3));
    float rc = fminf(1.f, 10.f / (rn + 1e-6f));       // STATE_CLIP = 10
    *(float4*)(out + (size_t)node * DDIM + d) = make_float4(n0 * rc, n1 * rc, n2 * rc, n3 * rc);
}

// ---------------- host launcher ----------------
extern "C" void kernel_launch(void* const* d_in, const int* in_sizes, int n_in,
                              void* d_out, int out_size, void* d_ws, size_t ws_size,
                              hipStream_t stream) {
    const float* G      = (const float*)d_in[0];
    const float* lng    = (const float*)d_in[1];
    const float* lnb    = (const float*)d_in[2];
    const float* WQ2    = (const float*)d_in[3];
    const float* WK2    = (const float*)d_in[4];
    const float* WQ3    = (const float*)d_in[5];
    const float* WK3    = (const float*)d_in[6];
    const float* Ttau   = (const float*)d_in[7];
    const float* WQm    = (const float*)d_in[8];
    const float* WKm    = (const float*)d_in[9];
    const float* Bmem   = (const float*)d_in[10];
    const float* lam2r  = (const float*)d_in[11];
    const float* lam3r  = (const float*)d_in[12];
    const float* lammr  = (const float*)d_in[13];
    const float* b2r    = (const float*)d_in[14];
    const float* b3r    = (const float*)d_in[15];
    const float* bmr    = (const float*)d_in[16];
    const int*   c2     = (const int*)d_in[17];
    const int*   u2     = (const int*)d_in[18];
    const int*   mt     = (const int*)d_in[19];
    float* out = (float*)d_out;

    // carve workspace (256B aligned regions)
    char* ws = (char*)d_ws;
    size_t cur = 0;
    auto carve = [&](size_t bytes) -> void* {
        void* p = ws + cur;
        cur = (cur + bytes + 255) & ~(size_t)255;
        return p;
    };
    unsigned short* g_bf   = (unsigned short*)carve((size_t)N_NODES * DDIM * 2);
    float*          mu     = (float*)carve((size_t)N_NODES * 4);
    float*          rsig   = (float*)carve((size_t)N_NODES * 4);
    unsigned short* wcat   = (unsigned short*)carve((size_t)FTOT * DDIM * 2);
    unsigned short* wcatT  = (unsigned short*)carve((size_t)FTOT * DDIM * 2);
    float*          Km     = (float*)carve((size_t)NHEAD * KSLOT * HDIM * 4);
    float*          P      = (float*)carve((size_t)N_NODES * FTOT * 4);
    float*          dP     = (float*)carve((size_t)N_NODES * FTOT * 4);
    unsigned short* dP_bf  = (unsigned short*)carve((size_t)N_NODES * FTOT * 2);
    float*          l2     = (float*)carve((size_t)N_EDGES * NHEAD * 4);
    float*          l3     = (float*)carve((size_t)N_EDGES * NHEAD * 4);
    unsigned*       m2     = (unsigned*)carve((size_t)N_NODES * NHEAD * 4);
    float*          s2     = (float*)carve((size_t)N_NODES * NHEAD * 4);
    unsigned*       m3     = (unsigned*)carve((size_t)N_NODES * NHEAD * 4);
    float*          s3     = (float*)carve((size_t)N_NODES * NHEAD * 4);
    float*          dg     = (float*)carve((size_t)N_NODES * DDIM * 4);
    float*          grad   = (float*)carve((size_t)N_NODES * DDIM * 4);
    float*          gn     = (float*)carve(256);
    (void)ws_size; (void)n_in; (void)in_sizes; (void)out_size;

    // zero the big scatter target (graph-capture safe)
    hipMemsetAsync(dP, 0, (size_t)N_NODES * FTOT * 4, stream);

    // weight prep
    k_build_wcat<<<(FTOT * DDIM + 255) / 256, 256, 0, stream>>>(WQ2, WK2, WQ3, WK3, WQm,
                                                                wcat, wcatT, FTOT * DDIM);
    k_build_km<<<(NHEAD * KSLOT * HDIM + 255) / 256, 256, 0, stream>>>(Bmem, WKm, Km,
                                                                       NHEAD * KSLOT * HDIM);
    // segment accumulators + grad-norm scalar
    k_init<<<(N_NODES * NHEAD + 255) / 256, 256, 0, stream>>>(m2, s2, m3, s3, gn,
                                                              N_NODES * NHEAD);
    // LayerNorm forward
    k_ln_fwd<<<N_NODES / 8, 256, 0, stream>>>(G, lng, lnb, g_bf, mu, rsig, N_NODES);

    // forward projections: P = g @ Wcat^T  (bf16 WMMA, fp32 accumulate), 16x64 tiles
    {
        dim3 grid(N_NODES / 16, FTOT / 64);
        k_gemm_bf16<<<grid, 32, 0, stream>>>((const bf16*)g_bf, DDIM,
                                             (const bf16*)wcat, DDIM,
                                             P, FTOT, DDIM);
    }

    // edge logits + segmented softmax stats
    k_edge_logits<<<(N_EDGES * NHEAD + 255) / 256, 256, 0, stream>>>(P, c2, u2, mt, Ttau,
                                                                     l2, l3, N_EDGES * NHEAD);
    k_segmax<<<(N_EDGES * NHEAD + 255) / 256, 256, 0, stream>>>(l2, l3, u2, m2, m3,
                                                                b2r, b3r, N_EDGES * NHEAD);
    k_segsum<<<(N_EDGES * NHEAD + 255) / 256, 256, 0, stream>>>(l2, l3, u2, m2, m3, s2, s3,
                                                                b2r, b3r, N_EDGES * NHEAD);

    // backward into dP
    k_mem_bwd<<<(N_NODES * NHEAD + 255) / 256, 256, 0, stream>>>(P, Km, dP, lammr, bmr,
                                                                 N_NODES * NHEAD);
    k_edge_bwd<<<(N_EDGES * NHEAD * HDIM + 255) / 256, 256, 0, stream>>>(
        P, l2, l3, m2, s2, m3, s3, c2, u2, mt, Ttau, dP,
        lam2r, lam3r, b2r, b3r, N_EDGES * NHEAD * HDIM);

    // dg = dP @ Wcat  (via WcatT rows; bf16 WMMA), 16x64 tiles
    k_cvt_bf16<<<(N_NODES * FTOT + 255) / 256, 256, 0, stream>>>(dP, dP_bf, N_NODES * FTOT);
    {
        dim3 grid(N_NODES / 16, DDIM / 64);
        k_gemm_bf16<<<grid, 32, 0, stream>>>((const bf16*)dP_bf, FTOT,
                                             (const bf16*)wcatT, FTOT,
                                             dg, DDIM, FTOT);
    }

    // LayerNorm backward + global grad norm, then clipped update
    k_ln_bwd<<<N_NODES / 8, 256, 0, stream>>>(G, lng, mu, rsig, dg, grad, gn, N_NODES);
    k_update<<<N_NODES / 8, 256, 0, stream>>>(G, grad, gn, out, N_NODES);
}
